// DSNet_37941741092931
// MI455X (gfx1250) — matrix operand order repaired and checked by reading, not compiled, not run on hardware
//
#include <hip/hip_runtime.h>

// Problem constants (from reference): P=200, C=10, F=128, B=65536, NU=0.9
#define PROTOS  200
#define PPAD    208   // pad to 13 tiles of 16
#define NCLS    10
#define NFEAT   128
#define BROWS   128   // batch rows per block
#define NTHR    128   // 4 waves of 32
#define DSTRIDE 209   // odd stride -> conflict-free per-row column walk

typedef __attribute__((ext_vector_type(16))) __bf16 v16bf;
typedef __attribute__((ext_vector_type(8)))  float  v8f;

union Frag { v16bf v; float4 f[2]; };

__device__ __forceinline__ unsigned short f2bf(float f) {
  unsigned int u = __float_as_uint(f);
  u += 0x7FFFu + ((u >> 16) & 1u);        // round-to-nearest-even
  return (unsigned short)(u >> 16);
}

__global__ __launch_bounds__(NTHR, 2)     // 2 blocks/WGP: 149 KB LDS each
void dsnet_fused(const float* __restrict__ x, const float* __restrict__ w,
                 const float* __restrict__ xi, const float* __restrict__ eta,
                 const float* __restrict__ beta, float* __restrict__ out)
{
  __shared__ __align__(16) unsigned short xbf[BROWS][NFEAT];   // 32 KB
  __shared__ __align__(16) unsigned short wbf[PPAD][NFEAT];    // 52 KB
  __shared__ _Float16 distH[BROWS][DSTRIDE];                   // 52.25 KB (dist, then si)
  __shared__ float gammaS[PPAD], alphaS[PPAD], wsqS[PPAD];
  __shared__ float xnormS[BROWS];
  __shared__ float uS[PPAD][12];                               // u[0..9], sum(u) at [10]

  const int t = threadIdx.x;
  const long rowBase = (long)blockIdx.x * BROWS;

  // ---- stage w -> bf16 + per-prototype constants ----
  for (int p = t; p < PPAD; p += NTHR) {
    float s = 0.f;
    if (p < PROTOS) {
      #pragma unroll 4
      for (int k = 0; k < NFEAT; ++k) {
        float v = w[p * NFEAT + k];
        s += v * v;
        wbf[p][k] = f2bf(v);
      }
    } else {
      for (int k = 0; k < NFEAT; ++k) wbf[p][k] = 0;
    }
    wsqS[p] = s;
    float ev  = (p < PROTOS) ? eta[p] : 0.f;
    gammaS[p] = ev * ev;
    float xiv = (p < PROTOS) ? xi[p] : 0.f;
    alphaS[p] = 1.f / (1.f + __expf(-xiv));
    float bsq[NCLS]; float bsum = 0.f;
    #pragma unroll
    for (int c = 0; c < NCLS; ++c) {
      float b = (p < PROTOS) ? beta[p * NCLS + c] : 0.f;
      bsq[c] = b * b; bsum += bsq[c];
    }
    float binv = 1.f / (bsum + 1e-8f);
    float su = 0.f;
    #pragma unroll
    for (int c = 0; c < NCLS; ++c) {
      float uu = (p < PROTOS) ? bsq[c] * binv : 0.f;
      uS[p][c] = uu; su += uu;
    }
    uS[p][10] = su;
  }

  // ---- stage x row t -> bf16 + exact f32 ||x||^2 ----
  {
    const float* xr = x + (rowBase + t) * NFEAT;
    float s = 0.f;
    #pragma unroll 4
    for (int k = 0; k < NFEAT; ++k) {
      float v = xr[k];
      s += v * v;
      xbf[t][k] = f2bf(v);
    }
    xnormS[t] = s;
  }
  __syncthreads();

  // ---- phase 1: dist = ||x||^2 + ||w||^2 - 2 x.w via v_wmma_f32_16x16x32_bf16 ----
  const int wv = t >> 5, lane = t & 31;
  const int m16 = lane & 15, h = lane >> 4;
  for (int tile = wv; tile < 8 * 13; tile += 4) {
    const int rt = tile / 13, ct = tile % 13;
    const int arow = rt * 16 + m16;     // A: lane holds row M = lane%16
    const int bcol = ct * 16 + m16;     // B: lane holds col N = lane%16
    v8f acc = {0.f, 0.f, 0.f, 0.f, 0.f, 0.f, 0.f, 0.f};
    #pragma unroll
    for (int kk = 0; kk < 4; ++kk) {    // K = 128 in 4 steps of 32
      const int kb = kk * 32;
      Frag a, b;
      // A 16-bit layout: elems 0..7 -> K = kb + 8h + 0..7 ; elems 8..15 -> K = kb+16+8h+0..7
      a.f[0] = *(const float4*)&xbf[arow][kb + 8 * h];
      a.f[1] = *(const float4*)&xbf[arow][kb + 16 + 8 * h];
      // B 16-bit layout: elems 0..7 -> K = kb + 8h + 0..7 ; elems 8..15 -> K = kb+16+8h+0..7
      b.f[0] = *(const float4*)&wbf[bcol][kb + 8 * h];
      b.f[1] = *(const float4*)&wbf[bcol][kb + 16 + 8 * h];
      acc = __builtin_amdgcn_wmma_f32_16x16x32_bf16(
              false, a.v, false, b.v, (short)0, acc, false, false);
    }
    // C/D layout: VGPR r, lane -> M = r + 8h, N = lane%16
    #pragma unroll
    for (int r = 0; r < 8; ++r) {
      const int drw = rt * 16 + 8 * h + r;
      distH[drw][bcol] = (_Float16)(xnormS[drw] + wsqS[bcol] - 2.f * acc[r]);
    }
  }
  __syncthreads();

  // ---- phase 2: one thread per batch row ----
  // pass A: si = alpha * exp(-gamma * dist); track row max in f32 (si > 0 always)
  float mx = 0.f;
  for (int p = 0; p < PROTOS; ++p) {
    float d = (float)distH[t][p];
    float s = alphaS[p] * __expf(-gammaS[p] * d);
    distH[t][p] = (_Float16)s;            // reuse dist row as si storage
    mx = fmaxf(mx, s);
  }
  const float inv = 1.f / (mx + 1e-4f);

  // Dempster scan over prototypes (matches reference column-wise formula,
  // including comb_omega = 3*m1w*m2w from applying all three terms to col C)
  float m[NCLS + 1];
  {
    float s0 = (float)distH[t][0] * inv;
    #pragma unroll
    for (int c = 0; c < NCLS; ++c) m[c] = s0 * uS[0][c];
    m[NCLS] = 1.f - s0 * uS[0][10];
  }
  for (int p = 1; p < PROTOS; ++p) {
    float sp  = (float)distH[t][p] * inv;
    float m2w = 1.f - sp * uS[p][10];
    float comb[NCLS + 1]; float tot = 0.f;
    #pragma unroll
    for (int c = 0; c < NCLS; ++c) {
      float m2 = sp * uS[p][c];
      comb[c] = m[c] * m2 + m[c] * m2w + m[NCLS] * m2;
      tot += comb[c];
    }
    comb[NCLS] = 3.f * m[NCLS] * m2w;
    tot += comb[NCLS];
    float nrm = 1.f / (tot + 1e-8f);
    #pragma unroll
    for (int c = 0; c <= NCLS; ++c) m[c] = comb[c] * nrm;
  }

  // DS3_normalize + DM redistribution
  float tot = 0.f;
  #pragma unroll
  for (int c = 0; c <= NCLS; ++c) tot += m[c];
  float nrm = 1.f / (tot + 1e-8f);
  #pragma unroll
  for (int c = 0; c <= NCLS; ++c) m[c] *= nrm;

  float* o = out + (rowBase + t) * NCLS;
  const float ig = (1.0f - 0.9f) * m[NCLS];
  #pragma unroll
  for (int c = 0; c < NCLS; ++c) o[c] = m[c] + ig;
}

extern "C" void kernel_launch(void* const* d_in, const int* in_sizes, int n_in,
                              void* d_out, int out_size, void* d_ws, size_t ws_size,
                              hipStream_t stream) {
  (void)in_sizes; (void)n_in; (void)d_ws; (void)ws_size; (void)out_size;
  const float* x    = (const float*)d_in[0];
  const float* w    = (const float*)d_in[1];
  const float* xi   = (const float*)d_in[2];
  const float* eta  = (const float*)d_in[3];
  const float* beta = (const float*)d_in[4];
  float* out = (float*)d_out;
  const int B = 65536;
  dim3 grid(B / BROWS), block(NTHR);
  dsnet_fused<<<grid, block, 0, stream>>>(x, w, xi, eta, beta, out);
}
